// GAT_LSTM_52905407152166
// MI455X (gfx1250) — matrix operand order repaired and checked by reading, not compiled, not run on hardware
//
#include <hip/hip_runtime.h>
#include <hip/hip_bf16.h>

// Dims from the reference
#define Hdim 128
#define Idim 64
#define Ndim 4096
#define Tdim 256
#define WROW 136   // padded LDS row stride (halves): 272B rows -> lane*68 dwords -> conflict-free half-waves

typedef _Float16 v16h __attribute__((ext_vector_type(16)));
typedef _Float16 v8h  __attribute__((ext_vector_type(8)));
typedef float    v8f  __attribute__((ext_vector_type(8)));

// ---- CDNA5 hardware tanh (TRANS pipe). v_nop covers the TRANS RAW hazard. ----
__device__ __forceinline__ float tanh_hw(float x) {
    float r;
    asm volatile("v_tanh_f32 %0, %1\n\tv_nop" : "=v"(r) : "v"(x));
    return r;
}
__device__ __forceinline__ float sigmoid_hw(float x) {
    return 0.5f * tanh_hw(0.5f * x) + 0.5f;
}

// ---- Kernel 1: convert 4x [H,I] fp32 weights to f16, k-major: wt[(g*64+k)*128 + h] ----
__global__ void wconv_kernel(const float* __restrict__ w0, const float* __restrict__ w1,
                             const float* __restrict__ w2, const float* __restrict__ w3,
                             _Float16* __restrict__ wt) {
    int idx = blockIdx.x * 256 + threadIdx.x;   // 0 .. 4*64*128-1
    if (idx >= 4 * Idim * Hdim) return;
    int g = idx >> 13;            // /8192
    int k = (idx >> 7) & 63;
    int h = idx & 127;
    const float* w = (g == 0) ? w0 : (g == 1) ? w1 : (g == 2) ? w2 : w3;
    wt[idx] = (_Float16)w[h * Idim + k];
}

// ---- Kernel 2: base_g[n][h] = sum_k h0[n,k]*wh_g[h,k] + b_i[h] + b_h[h] ----
__global__ void base_kernel(const float* __restrict__ hstate,
                            const float* __restrict__ wh0, const float* __restrict__ wh1,
                            const float* __restrict__ wh2, const float* __restrict__ wh3,
                            const float* __restrict__ bi0, const float* __restrict__ bi1,
                            const float* __restrict__ bi2, const float* __restrict__ bi3,
                            const float* __restrict__ bh0, const float* __restrict__ bh1,
                            const float* __restrict__ bh2, const float* __restrict__ bh3,
                            float* __restrict__ basef) {
    __shared__ float h0row[Hdim];
    int n = blockIdx.x;
    int h = threadIdx.x;
    h0row[h] = hstate[h * Ndim + n];   // h is [1,H,N]: h0[n,k] = hstate[k*N+n]
    __syncthreads();
    const float* whs[4] = {wh0, wh1, wh2, wh3};
    const float* bis[4] = {bi0, bi1, bi2, bi3};
    const float* bhs[4] = {bh0, bh1, bh2, bh3};
#pragma unroll
    for (int g = 0; g < 4; ++g) {
        const float* w = whs[g] + h * Hdim;
        float s = 0.f;
#pragma unroll 8
        for (int k = 0; k < Hdim; ++k) s += h0row[k] * w[k];
        basef[((size_t)g * Ndim + n) * Hdim + h] = s + bis[g][h] + bhs[g][h];
    }
}

// pack 8 fp32 -> halves into v16h at element offset 'base'
__device__ __forceinline__ void cvt8(v16h& d, int base, float4 a, float4 b) {
    d[base + 0] = (_Float16)a.x; d[base + 1] = (_Float16)a.y;
    d[base + 2] = (_Float16)a.z; d[base + 3] = (_Float16)a.w;
    d[base + 4] = (_Float16)b.x; d[base + 5] = (_Float16)b.y;
    d[base + 6] = (_Float16)b.z; d[base + 7] = (_Float16)b.w;
}

// strided gather along K (rows of WROW) from LDS -> v16h B operand: K {e, 16+e}
__device__ __forceinline__ v16h loadB(const _Float16* p) {
    v16h r;
#pragma unroll
    for (int e = 0; e < 8; ++e) { r[e] = p[e * WROW]; r[e + 8] = p[(16 + e) * WROW]; }
    return r;
}

// ---- Kernel 3: fused gate GEMM (WMMA f16) + activations + scatter ----
// Each wave: one 16-row tile (rows = (n,t) pairs; all 16 rows share one n since T=256).
__global__ __launch_bounds__(256) void lstm_main(const float* __restrict__ x,
                                                 const float* __restrict__ cstate,
                                                 const float* __restrict__ basef,
                                                 const _Float16* __restrict__ wt,
                                                 float* __restrict__ out) {
    __shared__ _Float16 wlds[4 * Idim * WROW];   // ~68KB, padded rows

    // cooperative weight staging: 4*64 rows of 128 halves -> padded LDS rows
    for (int i = threadIdx.x; i < 4 * Idim * 16; i += 256) {   // 16B chunks
        int row = i >> 4;
        int c = (i & 15) * 8;
        *(v8h*)(&wlds[row * WROW + c]) = *(const v8h*)(wt + row * 128 + c);
    }
    __syncthreads();

    const int lane  = threadIdx.x & 31;
    const int wave  = threadIdx.x >> 5;
    const int laneM = lane & 15;
    const int kgrp  = lane >> 4;

    const size_t rt = (size_t)blockIdx.x * 8 + wave;  // 16-row tile index
    const size_t r0 = rt * 16;
    const int n  = (int)(r0 >> 8);     // row = n*T + t, T = 256
    const int t0 = (int)(r0 & 255);

    // ---- A operands: 16x32 f16 layout. lane<16: K {kg*8..+7, 16+kg*8..}; +32 for a1 ----
    const float4* xr = (const float4*)(x + (r0 + (size_t)laneM) * Idim);
    const int kb = kgrp * 2;
    v16h a0, a1;
    {
        float4 f0 = xr[kb + 0],  f1 = xr[kb + 1];    // K = kgrp*8 .. +7
        float4 f2 = xr[kb + 4],  f3 = xr[kb + 5];    // K = 16+kgrp*8 ..
        float4 f4 = xr[kb + 8],  f5 = xr[kb + 9];    // K = 32+kgrp*8 ..
        float4 f6 = xr[kb + 12], f7 = xr[kb + 13];   // K = 48+kgrp*8 ..
        cvt8(a0, 0, f0, f1); cvt8(a0, 8, f2, f3);
        cvt8(a1, 0, f4, f5); cvt8(a1, 8, f6, f7);
    }

    const size_t NH = (size_t)Ndim * Hdim;
    const size_t o1 = (size_t)Tdim * NH;             // h_next offset
    const size_t o2 = o1 + (size_t)Hdim * Ndim;      // c_next offset

    for (int ct = 0; ct < 8; ++ct) {                 // 8 column tiles of H
        const int h = ct * 16 + laneM;               // this lane's output column
        const float c0v = cstate[(size_t)h * Ndim + n];

        v8f acc[4];
#pragma unroll
        for (int g = 0; g < 4; ++g) {                // 0=i 1=f 2=o 3=g (input order)
            float bv = basef[((size_t)g * Ndim + n) * Hdim + h];
            v8f c = {bv, bv, bv, bv, bv, bv, bv, bv};
            // B 32x16: lane = N(h), elements = K {kgrp*8..+7, 16+kgrp*8..}
            v16h b0 = loadB(&wlds[(g * 64 + kgrp * 8) * WROW + ct * 16 + laneM]);
            c = __builtin_amdgcn_wmma_f32_16x16x32_f16(false, a0, false, b0,
                                                       (short)0, c, false, false);
            v16h b1 = loadB(&wlds[(g * 64 + 32 + kgrp * 8) * WROW + ct * 16 + laneM]);
            c = __builtin_amdgcn_wmma_f32_16x16x32_f16(false, a1, false, b1,
                                                       (short)0, c, false, false);
            acc[g] = c;
        }

#pragma unroll
        for (int m = 0; m < 8; ++m) {                // C layout: M = m + 8*kgrp
            float iv = sigmoid_hw(acc[0][m]);
            float fv = sigmoid_hw(acc[1][m]);
            float ov = sigmoid_hw(acc[2][m]);
            float gv = tanh_hw(acc[3][m]);
            float cs = fv * c0v + iv * gv;
            float hs = ov * tanh_hw(cs);
            int t = t0 + m + 8 * kgrp;
            out[(size_t)t * NH + (size_t)n * Hdim + h] = hs;  // hidden_seq [T,N,H]
            if (t == Tdim - 1) {                               // h_next/c_next [H,N]
                out[o1 + (size_t)h * Ndim + n] = hs;
                out[o2 + (size_t)h * Ndim + n] = cs;
            }
        }
    }
}

// Workspace layout: [0, 8MB) = base (4*N*H f32), [8MB, +64KB) = wT f16
extern "C" void kernel_launch(void* const* d_in, const int* in_sizes, int n_in,
                              void* d_out, int out_size, void* d_ws, size_t ws_size,
                              hipStream_t stream) {
    const float* x   = (const float*)d_in[0];
    const float* hst = (const float*)d_in[1];
    const float* cst = (const float*)d_in[2];
    // dict order: gate 0=i (3..6), 1=f (7..10), 2=o (11..14), 3=g (15..18)
    float* basef = (float*)d_ws;
    _Float16* wt = (_Float16*)((char*)d_ws + (size_t)4 * Ndim * Hdim * sizeof(float));

    wconv_kernel<<<(4 * Idim * Hdim + 255) / 256, 256, 0, stream>>>(
        (const float*)d_in[3], (const float*)d_in[7],
        (const float*)d_in[11], (const float*)d_in[15], wt);

    base_kernel<<<Ndim, Hdim, 0, stream>>>(
        hst,
        (const float*)d_in[4],  (const float*)d_in[8],
        (const float*)d_in[12], (const float*)d_in[16],
        (const float*)d_in[5],  (const float*)d_in[9],
        (const float*)d_in[13], (const float*)d_in[17],
        (const float*)d_in[6],  (const float*)d_in[10],
        (const float*)d_in[14], (const float*)d_in[18],
        basef);

    // N*T/ (8 waves * 16 rows) = 8192 blocks
    lstm_main<<<(Ndim * Tdim) / 128, 256, 0, stream>>>(x, cst, basef, wt, (float*)d_out);
}
